// MultiHeadAttention_76871324663964
// MI455X (gfx1250) — matrix operand order repaired and checked by reading, not compiled, not run on hardware
//
#include <hip/hip_runtime.h>
#include <hip/hip_bf16.h>
#include <math.h>

typedef __attribute__((ext_vector_type(16))) __bf16 v16bf;
typedef __attribute__((ext_vector_type(8)))  float  v8f;
typedef __attribute__((ext_vector_type(2)))  __bf16 bf16x2;

#define BATCH 2
#define SEQ   2048
#define DIM   1024
#define HEADS 16
#define LDIM  64
#define MROWS (BATCH*SEQ)   // 4096

// ---------------------------------------------------------------------------
// 1) f32 -> bf16 convert
// ---------------------------------------------------------------------------
__global__ __launch_bounds__(256) void cvt_f32_bf16(const float* __restrict__ src,
                                                    __bf16* __restrict__ dst, int n) {
  int i = blockIdx.x * 256 + threadIdx.x;
  if (i < n) dst[i] = (__bf16)src[i];
}

// 2) transpose + convert weight: W[K][N] f32 -> WT[N][K] bf16
__global__ __launch_bounds__(256) void cvt_transpose_w(const float* __restrict__ W,
                                                       __bf16* __restrict__ WT,
                                                       int K, int N) {
  int i = blockIdx.x * 256 + threadIdx.x;
  if (i >= K * N) return;
  int n = i / K, k = i - n * K;
  WT[(size_t)n * K + k] = (__bf16)W[(size_t)k * N + n];
}

// ---------------------------------------------------------------------------
// 3) bf16 WMMA GEMM, batched over blockIdx.z (up to 3 weight/bias/output sets).
//    Wave computes a 32x64 strip (2 M-tiles x 4 N-tiles = 8 WMMAs / K-step);
//    block = 8 waves = 256x64 tile. K-loop double-buffered in registers.
// ---------------------------------------------------------------------------
struct GemmArgs {
  const __bf16* Bt[3];   // transposed weights [N][K]
  const float*  bias[3];
  __bf16*       outB[3]; // bf16 output (or null)
  float*        outF[3]; // f32 output (or null)
};

__global__ __launch_bounds__(256) void gemm_bf16_wmma(const __bf16* __restrict__ A,
                                                      GemmArgs args,
                                                      int M, int N, int K) {
  const int z    = blockIdx.z;
  const int lane = threadIdx.x & 31;
  const int wave = threadIdx.x >> 5;
  const int half = lane >> 4;
  const int l16  = lane & 15;
  const int m0   = blockIdx.x * 256 + wave * 32;
  const int n0   = blockIdx.y * 64;

  const __bf16* __restrict__ BT = args.Bt[z];
  const __bf16* aRow0 = A + (size_t)(m0 + l16) * K;
  const __bf16* aRow1 = A + (size_t)(m0 + 16 + l16) * K;
  const __bf16* bRow0 = BT + (size_t)(n0 + l16) * K;

  auto loadA = [&](v16bf& af, const __bf16* row, int kc) {
#pragma unroll
    for (int v = 0; v < 8; ++v) {
      int k = kc + 2 * (v & 3) + ((v >> 2) << 4) + half * 8;   // A 16x32 layout
      bf16x2 pr = *(const bf16x2*)(row + k);
      af[2 * v] = pr.x; af[2 * v + 1] = pr.y;
    }
  };
  auto loadB = [&](v16bf& bfv, int t, int kc) {
#pragma unroll
    for (int v = 0; v < 8; ++v) {
      int k = kc + 2 * (v & 3) + ((v >> 2) << 4) + half * 8;   // B 32x16 layout
      bf16x2 pr = *(const bf16x2*)(bRow0 + (size_t)(16 * t) * K + k);
      bfv[2 * v] = pr.x; bfv[2 * v + 1] = pr.y;
    }
  };

  v8f acc[2][4] = {};
  v16bf a0[2], a1[2], b0[4], b1[4];

  // prologue: fragments for kc = 0
  loadA(a0[0], aRow0, 0); loadA(a0[1], aRow1, 0);
#pragma unroll
  for (int t = 0; t < 4; ++t) loadB(b0[t], t, 0);

  for (int kc = 0; kc < K; kc += 64) {
    // stage next 32-chunk while set0 feeds the WMMAs
    loadA(a1[0], aRow0, kc + 32); loadA(a1[1], aRow1, kc + 32);
#pragma unroll
    for (int t = 0; t < 4; ++t) loadB(b1[t], t, kc + 32);
    if (kc + 128 < K) {
      __builtin_prefetch(aRow0 + kc + 128, 0, 0);
      __builtin_prefetch(aRow1 + kc + 128, 0, 0);
    }
#pragma unroll
    for (int t = 0; t < 4; ++t) {
      acc[0][t] = __builtin_amdgcn_wmma_f32_16x16x32_bf16(false, a0[0], false, b0[t],
                                                          (short)0, acc[0][t], false, false);
      acc[1][t] = __builtin_amdgcn_wmma_f32_16x16x32_bf16(false, a0[1], false, b0[t],
                                                          (short)0, acc[1][t], false, false);
    }
    if (kc + 64 < K) {   // stage set0 for the next iteration
      loadA(a0[0], aRow0, kc + 64); loadA(a0[1], aRow1, kc + 64);
#pragma unroll
      for (int t = 0; t < 4; ++t) loadB(b0[t], t, kc + 64);
    }
#pragma unroll
    for (int t = 0; t < 4; ++t) {
      acc[0][t] = __builtin_amdgcn_wmma_f32_16x16x32_bf16(false, a1[0], false, b1[t],
                                                          (short)0, acc[0][t], false, false);
      acc[1][t] = __builtin_amdgcn_wmma_f32_16x16x32_bf16(false, a1[1], false, b1[t],
                                                          (short)0, acc[1][t], false, false);
    }
  }

  const float* bias = args.bias[z];
  __bf16* outB = args.outB[z];
  float*  outF = args.outF[z];
#pragma unroll
  for (int t = 0; t < 4; ++t) {
    float bval = bias ? bias[n0 + 16 * t + l16] : 0.0f;
#pragma unroll
    for (int mt = 0; mt < 2; ++mt) {
#pragma unroll
      for (int r = 0; r < 8; ++r) {
        float v = acc[mt][t][r] + bval;
        int row = m0 + 16 * mt + r + 8 * half;   // D layout: M = r + 8*half
        int col = n0 + 16 * t + l16;             // N = lane&15 (+16t)
        if (outF) outF[(size_t)row * N + col] = v;
        else      outB[(size_t)row * N + col] = (__bf16)v;
      }
    }
  }
}

// ---------------------------------------------------------------------------
// 4) RoPE + transpose: raw [B,S,H,L] -> rot [B,H,S,L]
// ---------------------------------------------------------------------------
__global__ __launch_bounds__(256) void rope_rot(const __bf16* __restrict__ raw,
                                                __bf16* __restrict__ rot) {
  int i = blockIdx.x * 256 + threadIdx.x;      // B*S*H*32 = 2,097,152 threads
  if (i >= BATCH * SEQ * HEADS * 32) return;
  int j = i & 31;
  int h = (i >> 5) & 15;
  int s = (i >> 9) & 2047;
  int b = i >> 20;
  size_t ibase = (((size_t)b * SEQ + s) * HEADS + h) * LDIM;
  size_t obase = (((size_t)b * HEADS + h) * SEQ + s) * LDIM;
  float t0 = (float)raw[ibase + 2 * j];
  float t1 = (float)raw[ibase + 2 * j + 1];
  float inv = expf(-(float)j * 0.28782313662425574f);   // 10000^(-j/32)
  float ang = (float)s * inv;
  float c = cosf(ang), si = sinf(ang);
  rot[obase + j]      = (__bf16)(t0 * c - t1 * si);
  rot[obase + 32 + j] = (__bf16)(t0 * si + t1 * c);
}

// 5) V transpose: raw [B,S,H,L] -> vt [B,H,L,S]
__global__ __launch_bounds__(256) void v_transpose(const __bf16* __restrict__ raw,
                                                   __bf16* __restrict__ vt) {
  int i = blockIdx.x * 256 + threadIdx.x;      // 4,194,304 threads
  if (i >= BATCH * SEQ * HEADS * LDIM) return;
  int l = i & 63;
  int h = (i >> 6) & 15;
  int s = (i >> 10) & 2047;
  int b = i >> 21;
  vt[(((size_t)b * HEADS + h) * LDIM + l) * SEQ + s] = raw[i];
}

// ---------------------------------------------------------------------------
// 6) Causal flash attention. One wave per 16-query tile. kv blocks of 32.
//    Q,K in [B,H,S,L] (bf16), V in [B,H,L,S] (bf16), out att [B,S,H*L] (bf16).
// ---------------------------------------------------------------------------
__global__ __launch_bounds__(256) void attn_flash(const __bf16* __restrict__ Q,
                                                  const __bf16* __restrict__ Kr,
                                                  const __bf16* __restrict__ Vt,
                                                  __bf16* __restrict__ Att) {
  __shared__ __bf16 ldsP[8 * 16 * 32];         // 8 KB: per-wave 16x32 prob tile
  const int lane = threadIdx.x & 31;
  const int wave = threadIdx.x >> 5;
  const int tile = blockIdx.x * 8 + wave;      // 4096 tiles total
  const int qt = tile & 127;
  const int h  = (tile >> 7) & 15;
  const int b  = tile >> 11;
  const int q0 = qt * 16;
  const int half = lane >> 4;
  const int l16  = lane & 15;

  const __bf16* qbase = Q  + (((size_t)b * HEADS + h) * SEQ + q0) * LDIM;
  const __bf16* kbase = Kr + (((size_t)b * HEADS + h) * SEQ) * LDIM;
  const __bf16* vbase = Vt + (((size_t)b * HEADS + h) * LDIM) * SEQ;
  __bf16* lp = &ldsP[wave * 16 * 32];

  v16bf qf[2];
#pragma unroll
  for (int kc = 0; kc < 2; ++kc)
#pragma unroll
    for (int v = 0; v < 8; ++v) {
      int k = kc * 32 + 2 * (v & 3) + ((v >> 2) << 4) + half * 8;
      bf16x2 pr = *(const bf16x2*)(qbase + (size_t)l16 * LDIM + k);
      qf[kc][2 * v] = pr.x; qf[kc][2 * v + 1] = pr.y;
    }

  v8f accO[4] = {};
  float m[8], lsum[8];
#pragma unroll
  for (int r = 0; r < 8; ++r) { m[r] = -3.0e38f; lsum[r] = 0.0f; }

  const int nb = (q0 + 16 + 31) >> 5;
  for (int kb = 0; kb < nb; ++kb) {
    const int kv0 = kb * 32;
    float p[2][8], mnew[8];
#pragma unroll
    for (int r = 0; r < 8; ++r) mnew[r] = m[r];

#pragma unroll
    for (int j = 0; j < 2; ++j) {
      v8f sc = {};
#pragma unroll
      for (int kc = 0; kc < 2; ++kc) {
        v16bf bfv;
#pragma unroll
        for (int v = 0; v < 8; ++v) {
          int kl = kc * 32 + 2 * (v & 3) + ((v >> 2) << 4) + half * 8;
          bf16x2 pr = *(const bf16x2*)(kbase + (size_t)(kv0 + j * 16 + l16) * LDIM + kl);
          bfv[2 * v] = pr.x; bfv[2 * v + 1] = pr.y;
        }
        sc = __builtin_amdgcn_wmma_f32_16x16x32_bf16(false, qf[kc], false, bfv,
                                                     (short)0, sc, false, false);
      }
      int key = kv0 + j * 16 + l16;
#pragma unroll
      for (int r = 0; r < 8; ++r) {
        float sv = sc[r] * 0.125f;             // 1/sqrt(64)
        int q = q0 + r + 8 * half;
        sv = (key <= q) ? sv : -1.0e30f;       // causal mask
        p[j][r] = sv;
        mnew[r] = fmaxf(mnew[r], sv);
      }
    }
#pragma unroll
    for (int r = 0; r < 8; ++r) {
      float v = mnew[r];
      v = fmaxf(v, __shfl_xor(v, 1, 16));
      v = fmaxf(v, __shfl_xor(v, 2, 16));
      v = fmaxf(v, __shfl_xor(v, 4, 16));
      v = fmaxf(v, __shfl_xor(v, 8, 16));
      mnew[r] = v;
    }
#pragma unroll
    for (int r = 0; r < 8; ++r) {
      float alpha = __expf(m[r] - mnew[r]);
      m[r] = mnew[r];
      p[0][r] = __expf(p[0][r] - mnew[r]);
      p[1][r] = __expf(p[1][r] - mnew[r]);
      float ps = p[0][r] + p[1][r];
      ps += __shfl_xor(ps, 1, 16);
      ps += __shfl_xor(ps, 2, 16);
      ps += __shfl_xor(ps, 4, 16);
      ps += __shfl_xor(ps, 8, 16);
      lsum[r] = lsum[r] * alpha + ps;
#pragma unroll
      for (int t = 0; t < 4; ++t) accO[t][r] *= alpha;
    }
#pragma unroll
    for (int r = 0; r < 8; ++r) {
      int row = r + 8 * half;
      lp[row * 32 + l16]      = (__bf16)p[0][r];
      lp[row * 32 + 16 + l16] = (__bf16)p[1][r];
    }
    v16bf pf;
#pragma unroll
    for (int v = 0; v < 8; ++v) {
      int k = 2 * (v & 3) + ((v >> 2) << 4) + half * 8;
      bf16x2 pr = *(const bf16x2*)(lp + l16 * 32 + k);
      pf[2 * v] = pr.x; pf[2 * v + 1] = pr.y;
    }
#pragma unroll
    for (int t = 0; t < 4; ++t) {
      v16bf vf;
#pragma unroll
      for (int v = 0; v < 8; ++v) {
        int kv = 2 * (v & 3) + ((v >> 2) << 4) + half * 8;
        bf16x2 pr = *(const bf16x2*)(vbase + (size_t)(16 * t + l16) * SEQ + kv0 + kv);
        vf[2 * v] = pr.x; vf[2 * v + 1] = pr.y;
      }
      accO[t] = __builtin_amdgcn_wmma_f32_16x16x32_bf16(false, pf, false, vf,
                                                        (short)0, accO[t], false, false);
    }
  }
  __bf16* abase = Att + ((size_t)b * SEQ + q0) * DIM + h * LDIM;
#pragma unroll
  for (int r = 0; r < 8; ++r) {
    float inv = 1.0f / lsum[r];
    int row = r + 8 * half;
#pragma unroll
    for (int t = 0; t < 4; ++t)
      abase[(size_t)row * DIM + 16 * t + l16] = (__bf16)(accO[t][r] * inv);
  }
}

// ---------------------------------------------------------------------------
// Host launcher
// ---------------------------------------------------------------------------
extern "C" void kernel_launch(void* const* d_in, const int* in_sizes, int n_in,
                              void* d_out, int out_size, void* d_ws, size_t ws_size,
                              hipStream_t stream) {
  (void)in_sizes; (void)n_in; (void)out_size; (void)ws_size;
  const float* x  = (const float*)d_in[0];
  const float* Wq = (const float*)d_in[1]; const float* bq = (const float*)d_in[2];
  const float* Wk = (const float*)d_in[3]; const float* bk = (const float*)d_in[4];
  const float* Wv = (const float*)d_in[5]; const float* bv = (const float*)d_in[6];
  const float* Wo = (const float*)d_in[7]; const float* bo = (const float*)d_in[8];
  float* out = (float*)d_out;

  char* ws = (char*)d_ws;
  const size_t MB = 1024ull * 1024ull;
  __bf16* xb   = (__bf16*)(ws + 0);          //  8 MB  [4096,1024]
  __bf16* WqT  = (__bf16*)(ws +  8 * MB);    //  2 MB  [1024,1024] transposed
  __bf16* WkT  = (__bf16*)(ws + 10 * MB);
  __bf16* WvT  = (__bf16*)(ws + 12 * MB);
  __bf16* WoT  = (__bf16*)(ws + 14 * MB);
  __bf16* Qraw = (__bf16*)(ws + 16 * MB);    //  8 MB  [B,S,H*L]  (reused for att)
  __bf16* Kraw = (__bf16*)(ws + 24 * MB);
  __bf16* Vraw = (__bf16*)(ws + 32 * MB);
  __bf16* Qrot = (__bf16*)(ws + 40 * MB);    //  8 MB  [B,H,S,L]
  __bf16* Krot = (__bf16*)(ws + 48 * MB);
  __bf16* VtT  = (__bf16*)(ws + 56 * MB);    //  8 MB  [B,H,L,S]
  __bf16* att  = Qraw;                       //  reuse after rope consumed Qraw

  const int NX = MROWS * DIM;                // 4,194,304
  cvt_f32_bf16<<<(NX + 255) / 256, 256, 0, stream>>>(x, xb, NX);
  const int NW = DIM * DIM;                  // 1,048,576
  cvt_transpose_w<<<(NW + 255) / 256, 256, 0, stream>>>(Wq, WqT, DIM, DIM);
  cvt_transpose_w<<<(NW + 255) / 256, 256, 0, stream>>>(Wk, WkT, DIM, DIM);
  cvt_transpose_w<<<(NW + 255) / 256, 256, 0, stream>>>(Wv, WvT, DIM, DIM);
  cvt_transpose_w<<<(NW + 255) / 256, 256, 0, stream>>>(Wo, WoT, DIM, DIM);

  // fused QKV projections: gridDim.z selects weight/bias/output
  GemmArgs qkv = {};
  qkv.Bt[0] = WqT; qkv.Bt[1] = WkT; qkv.Bt[2] = WvT;
  qkv.bias[0] = bq; qkv.bias[1] = bk; qkv.bias[2] = bv;
  qkv.outB[0] = Qraw; qkv.outB[1] = Kraw; qkv.outB[2] = Vraw;
  dim3 gq(MROWS / 256, DIM / 64, 3);         // 16 x 16 x 3
  gemm_bf16_wmma<<<gq, 256, 0, stream>>>(xb, qkv, MROWS, DIM, DIM);

  const int NR = BATCH * SEQ * HEADS * 32;   // 2,097,152
  rope_rot<<<(NR + 255) / 256, 256, 0, stream>>>(Qraw, Qrot);
  rope_rot<<<(NR + 255) / 256, 256, 0, stream>>>(Kraw, Krot);
  v_transpose<<<(NX + 255) / 256, 256, 0, stream>>>(Vraw, VtT);

  attn_flash<<<512, 256, 0, stream>>>(Qrot, Krot, VtT, att);

  // output projection -> f32 d_out
  GemmArgs oproj = {};
  oproj.Bt[0] = WoT; oproj.bias[0] = bo; oproj.outF[0] = out;
  dim3 go(MROWS / 256, DIM / 64, 1);
  gemm_bf16_wmma<<<go, 256, 0, stream>>>(att, oproj, MROWS, DIM, DIM);
}
